// ContextualSimilarity_43130061586992
// MI455X (gfx1250) — compile-verified, matrix operand order checked by reading, not run on hardware
//
#include <hip/hip_runtime.h>
#include <hip/hip_bf16.h>

#define NV 8192
#define DV 512

typedef __attribute__((ext_vector_type(16))) _Float16 v16h;
typedef __attribute__((ext_vector_type(8)))  _Float16 v8h;
typedef __attribute__((ext_vector_type(8)))  float    v8f;

// ---------- helpers ----------

__device__ __forceinline__ v8f wmma_f16(v16h a, v16h b, v8f c) {
  // (neg_a, A, neg_b, B, c_mod, C, reuse_a, reuse_b)
  return __builtin_amdgcn_wmma_f32_16x16x32_f16(false, a, false, b, (short)0, c, false, false);
}

// Load one 16x32 f16 WMMA A/B fragment from an LDS tile (pitch 40 halves).
// Documented 16-bit layout: lane = row (0..15) in low half-wave; lanes 16-31
// hold K+8; v16h elems 0..7 = K[kh..kh+7], elems 8..15 = K[16+kh..16+kh+7].
__device__ __forceinline__ v16h frag_lds(const _Float16* base) {
  int l  = threadIdx.x & 31;
  int r  = l & 15;
  int kh = (l >> 4) << 3;          // 0 or 8
  const _Float16* p = base + r * 40 + kh;
  union { v16h v; struct { v8h a, b; } s; } u;
  u.s.a = *(const v8h*)(p);
  u.s.b = *(const v8h*)(p + 16);
  return u.v;
}

// Cooperative global->LDS copy of a rows x 32-half tile (dst pitch 40 halves).
__device__ __forceinline__ void load_tile_h(_Float16* dst, const _Float16* src,
                                            long long row0, long long ldg,
                                            long long k0, int rows) {
  for (int idx = threadIdx.x; idx < rows * 4; idx += 256) {
    int r = idx >> 2;
    int c = (idx & 3) << 3;
    *(v8h*)(dst + r * 40 + c) = *(const v8h*)(src + (row0 + r) * ldg + k0 + c);
  }
}

__device__ __forceinline__ float block_reduce_sum(float v, float* sbuf) {
  int t = threadIdx.x;
  sbuf[t] = v; __syncthreads();
  for (int s = 128; s > 0; s >>= 1) {
    if (t < s) sbuf[t] += sbuf[t + s];
    __syncthreads();
  }
  return sbuf[0];
}

__device__ __forceinline__ void ins5(float (&t)[5], float v) {
  if (v < t[4]) {
    t[4] = v; float tmp;
    if (t[4] < t[3]) { tmp = t[3]; t[3] = t[4]; t[4] = tmp; }
    if (t[3] < t[2]) { tmp = t[2]; t[2] = t[3]; t[3] = tmp; }
    if (t[2] < t[1]) { tmp = t[1]; t[1] = t[2]; t[2] = tmp; }
    if (t[1] < t[0]) { tmp = t[0]; t[0] = t[1]; t[1] = tmp; }
  }
}

// ---------- 1) row norms + fp32->hi/lo f16 split ----------

__global__ __launch_bounds__(256) void prep_kernel(const float* __restrict__ z,
                                                   _Float16* __restrict__ zh,
                                                   _Float16* __restrict__ zl,
                                                   float* __restrict__ sq) {
  int row  = blockIdx.x * 8 + (threadIdx.x >> 5);
  int lane = threadIdx.x & 31;
  const float* zr = z + (long long)row * DV;
  float s = 0.f;
  for (int c = lane; c < DV; c += 32) {
    float v = zr[c];
    _Float16 h = (_Float16)v;
    zh[(long long)row * DV + c] = h;
    zl[(long long)row * DV + c] = (_Float16)(v - (float)h);
    s += v * v;
  }
  for (int o = 16; o > 0; o >>= 1) s += __shfl_down(s, o, 32);
  if (lane == 0) sq[row] = s;
}

// ---------- 2) distances via hi/lo split WMMA GEMM ----------
// block tile 128(M) x 64(N); 8 waves = 4x2 of 32x32 wave tiles; K-step 32.

__global__ __launch_bounds__(256) void dist_kernel(const _Float16* __restrict__ zh,
                                                   const _Float16* __restrict__ zl,
                                                   const float* __restrict__ sq,
                                                   float* __restrict__ dist) {
  __shared__ _Float16 sAh[128 * 40], sAl[128 * 40], sBh[64 * 40], sBl[64 * 40];
  const long long i0 = (long long)blockIdx.y * 128;
  const long long j0 = (long long)blockIdx.x * 64;
  const int wave = threadIdx.x >> 5;
  const int m0 = (wave & 3) * 32;
  const int n0 = (wave >> 2) * 32;
  v8f acc[2][2] = {};
  for (int kb = 0; kb < DV; kb += 32) {
    __syncthreads();
    load_tile_h(sAh, zh, i0, DV, kb, 128);
    load_tile_h(sAl, zl, i0, DV, kb, 128);
    load_tile_h(sBh, zh, j0, DV, kb, 64);
    load_tile_h(sBl, zl, j0, DV, kb, 64);
    __syncthreads();
    v16h ah[2], al[2], bh[2], bl[2];
#pragma unroll
    for (int q = 0; q < 2; ++q) {
      ah[q] = frag_lds(sAh + (m0 + q * 16) * 40);
      al[q] = frag_lds(sAl + (m0 + q * 16) * 40);
      bh[q] = frag_lds(sBh + (n0 + q * 16) * 40);
      bl[q] = frag_lds(sBl + (n0 + q * 16) * 40);
    }
#pragma unroll
    for (int mi = 0; mi < 2; ++mi)
#pragma unroll
      for (int ni = 0; ni < 2; ++ni) {
        acc[mi][ni] = wmma_f16(ah[mi], bh[ni], acc[mi][ni]);
        acc[mi][ni] = wmma_f16(ah[mi], bl[ni], acc[mi][ni]);
        acc[mi][ni] = wmma_f16(al[mi], bh[ni], acc[mi][ni]);
      }
  }
  const int l  = threadIdx.x & 31;
  const int ln = l & 15;
  const int lh = (l >> 4) << 3;
#pragma unroll
  for (int mi = 0; mi < 2; ++mi)
#pragma unroll
    for (int ni = 0; ni < 2; ++ni) {
      long long col = j0 + n0 + ni * 16 + ln;
      float sqc = sq[col];
#pragma unroll
      for (int r = 0; r < 8; ++r) {
        long long row = i0 + m0 + mi * 16 + lh + r;
        float g  = acc[mi][ni][r];
        float d2 = sq[row] + sqc - 2.f * g;
        dist[row * NV + col] = sqrtf(fmaxf(d2, 0.f));
      }
    }
}

// ---------- 3) 5th-smallest per row ----------

__global__ __launch_bounds__(256) void kth_kernel(const float* __restrict__ dist,
                                                  float* __restrict__ kth) {
  __shared__ float buf[256 * 5];
  __shared__ float buf2[32 * 5];
  const float* dr = dist + (long long)blockIdx.x * NV;
  float t[5] = {3.0e38f, 3.0e38f, 3.0e38f, 3.0e38f, 3.0e38f};
  for (int j = threadIdx.x; j < NV; j += 256) ins5(t, dr[j]);
#pragma unroll
  for (int s = 0; s < 5; ++s) buf[threadIdx.x * 5 + s] = t[s];
  __syncthreads();
  if (threadIdx.x < 32) {
    float m[5] = {3.0e38f, 3.0e38f, 3.0e38f, 3.0e38f, 3.0e38f};
    for (int q = 0; q < 40; ++q) ins5(m, buf[threadIdx.x * 40 + q]);
#pragma unroll
    for (int s = 0; s < 5; ++s) buf2[threadIdx.x * 5 + s] = m[s];
  }
  __syncthreads();
  if (threadIdx.x == 0) {
    float m[5] = {3.0e38f, 3.0e38f, 3.0e38f, 3.0e38f, 3.0e38f};
    for (int q = 0; q < 160; ++q) ins5(m, buf2[q]);
    kth[blockIdx.x] = m[4];
  }
}

// ---------- 4) mask (faithful column-threshold bug) + row sums ----------

__global__ __launch_bounds__(256) void mask_kernel(const float* __restrict__ dist,
                                                   const float* __restrict__ kth,
                                                   _Float16* __restrict__ maskh,
                                                   float* __restrict__ msum) {
  __shared__ float red[256];
  long long row = blockIdx.x;
  const float* dr = dist + row * NV;
  float cnt = 0.f;
  for (int j = threadIdx.x; j < NV; j += 256) {
    float m = (dr[j] <= kth[j]) ? 1.f : 0.f;   // compare vs column threshold
    maskh[row * NV + j] = (_Float16)m;
    cnt += m;
  }
  float s = block_reduce_sum(cnt, red);
  if (threadIdx.x == 0) msum[row] = s;
}

// ---------- 5) R = M .* M^T (64x64 tiles, transpose via LDS) ----------

__global__ __launch_bounds__(256) void r_kernel(const _Float16* __restrict__ maskh,
                                                _Float16* __restrict__ Rh) {
  __shared__ _Float16 tA[64 * 72], tB[64 * 72];
  long long i0 = (long long)blockIdx.y * 64;
  long long j0 = (long long)blockIdx.x * 64;
  for (int idx = threadIdx.x; idx < 64 * 8; idx += 256) {
    int r = idx >> 3, c = (idx & 7) << 3;
    *(v8h*)(tA + r * 72 + c) = *(const v8h*)(maskh + (i0 + r) * NV + j0 + c);
    *(v8h*)(tB + r * 72 + c) = *(const v8h*)(maskh + (j0 + r) * NV + i0 + c);
  }
  __syncthreads();
  for (int idx = threadIdx.x; idx < 64 * 64; idx += 256) {
    int r = idx >> 6, c = idx & 63;
    Rh[(i0 + r) * NV + j0 + c] = tA[r * 72 + c] * tB[c * 72 + r];
  }
}

__global__ __launch_bounds__(256) void rsum_kernel(const _Float16* __restrict__ Rh,
                                                   float* __restrict__ rsum) {
  __shared__ float red[256];
  long long row = blockIdx.x;
  float cnt = 0.f;
  for (int j = threadIdx.x; j < NV; j += 256) cnt += (float)Rh[row * NV + j];
  float s = block_reduce_sum(cnt, red);
  if (threadIdx.x == 0) rsum[row] = s;
}

// ---------- 6) S1 = (M M^T) / msum[row]  (exact f16 WMMA: M is 0/1) ----------

__global__ __launch_bounds__(256) void s1_kernel(const _Float16* __restrict__ maskh,
                                                 const float* __restrict__ msum,
                                                 float* __restrict__ S1) {
  __shared__ _Float16 sA[128 * 40], sB[64 * 40];
  const long long i0 = (long long)blockIdx.y * 128;
  const long long j0 = (long long)blockIdx.x * 64;
  const int wave = threadIdx.x >> 5;
  const int m0 = (wave & 3) * 32;
  const int n0 = (wave >> 2) * 32;
  v8f acc[2][2] = {};
  for (int kb = 0; kb < NV; kb += 32) {
    __syncthreads();
    load_tile_h(sA, maskh, i0, NV, kb, 128);
    load_tile_h(sB, maskh, j0, NV, kb, 64);
    __syncthreads();
    v16h a[2], b[2];
#pragma unroll
    for (int q = 0; q < 2; ++q) {
      a[q] = frag_lds(sA + (m0 + q * 16) * 40);
      b[q] = frag_lds(sB + (n0 + q * 16) * 40);
    }
#pragma unroll
    for (int mi = 0; mi < 2; ++mi)
#pragma unroll
      for (int ni = 0; ni < 2; ++ni)
        acc[mi][ni] = wmma_f16(a[mi], b[ni], acc[mi][ni]);
  }
  const int l  = threadIdx.x & 31;
  const int ln = l & 15;
  const int lh = (l >> 4) << 3;
#pragma unroll
  for (int mi = 0; mi < 2; ++mi)
#pragma unroll
    for (int ni = 0; ni < 2; ++ni) {
      long long col = j0 + n0 + ni * 16 + ln;
#pragma unroll
      for (int r = 0; r < 8; ++r) {
        long long row = i0 + m0 + mi * 16 + lh + r;
        S1[row * NV + col] = acc[mi][ni][r] * (1.f / msum[row]);
      }
    }
}

// ---------- 7) S2 = (S1 R^T) / rsum[row]  (S1 split hi/lo f16) ----------

__global__ __launch_bounds__(256) void s2_kernel(const float* __restrict__ S1,
                                                 const _Float16* __restrict__ Rh,
                                                 const float* __restrict__ rsum,
                                                 float* __restrict__ out) {
  __shared__ _Float16 sAh[128 * 40], sAl[128 * 40], sB[64 * 40];
  const long long i0 = (long long)blockIdx.y * 128;
  const long long j0 = (long long)blockIdx.x * 64;
  const int wave = threadIdx.x >> 5;
  const int m0 = (wave & 3) * 32;
  const int n0 = (wave >> 2) * 32;
  v8f acc[2][2] = {};
  for (int kb = 0; kb < NV; kb += 32) {
    __syncthreads();
    for (int idx = threadIdx.x; idx < 128 * 32; idx += 256) {
      int r = idx >> 5, c = idx & 31;
      float v = S1[(i0 + r) * NV + kb + c];
      _Float16 h = (_Float16)v;
      sAh[r * 40 + c] = h;
      sAl[r * 40 + c] = (_Float16)(v - (float)h);
    }
    load_tile_h(sB, Rh, j0, NV, kb, 64);
    __syncthreads();
    v16h ah[2], al[2], b[2];
#pragma unroll
    for (int q = 0; q < 2; ++q) {
      ah[q] = frag_lds(sAh + (m0 + q * 16) * 40);
      al[q] = frag_lds(sAl + (m0 + q * 16) * 40);
      b[q]  = frag_lds(sB  + (n0 + q * 16) * 40);
    }
#pragma unroll
    for (int mi = 0; mi < 2; ++mi)
#pragma unroll
      for (int ni = 0; ni < 2; ++ni) {
        acc[mi][ni] = wmma_f16(ah[mi], b[ni], acc[mi][ni]);
        acc[mi][ni] = wmma_f16(al[mi], b[ni], acc[mi][ni]);
      }
  }
  const int l  = threadIdx.x & 31;
  const int ln = l & 15;
  const int lh = (l >> 4) << 3;
#pragma unroll
  for (int mi = 0; mi < 2; ++mi)
#pragma unroll
    for (int ni = 0; ni < 2; ++ni) {
      long long col = j0 + n0 + ni * 16 + ln;
#pragma unroll
      for (int r = 0; r < 8; ++r) {
        long long row = i0 + m0 + mi * 16 + lh + r;
        out[row * NV + col] = acc[mi][ni][r] * (1.f / rsum[row]);
      }
    }
}

// ---------- 8) out = 0.5 * (out + out^T), in place by 64x64 tile pairs ----------

__global__ __launch_bounds__(256) void sym_kernel(float* __restrict__ out) {
  int bi = blockIdx.y, bj = blockIdx.x;
  if (bi > bj) return;
  __shared__ float tU[64 * 64], tL[64 * 64];
  long long i0 = (long long)bi * 64, j0 = (long long)bj * 64;
  for (int idx = threadIdx.x; idx < 64 * 64; idx += 256) {
    int r = idx >> 6, c = idx & 63;
    tU[idx] = out[(i0 + r) * NV + j0 + c];
    tL[idx] = out[(j0 + r) * NV + i0 + c];
  }
  __syncthreads();
  for (int idx = threadIdx.x; idx < 64 * 64; idx += 256) {
    int r = idx >> 6, c = idx & 63;
    float v = 0.5f * (tU[r * 64 + c] + tL[c * 64 + r]);
    out[(i0 + r) * NV + j0 + c] = v;
    if (bi != bj) out[(j0 + c) * NV + i0 + r] = v;
    else          out[(i0 + c) * NV + j0 + r] = 0.5f * (tU[c * 64 + r] + tU[r * 64 + c]);
  }
}

// ---------- launch ----------
// ws layout (bytes): dist/S1 f32 N*N | mask f16 N*N | R f16 N*N | zh f16 N*D |
//                    zl f16 N*D | sq,kth,msum,rsum f32 N each.
// Requires ws_size >= 8*N*N + 4*N*D + 16*N  (~529 MB for N=8192, D=512).

extern "C" void kernel_launch(void* const* d_in, const int* in_sizes, int n_in,
                              void* d_out, int out_size, void* d_ws, size_t ws_size,
                              hipStream_t stream) {
  const float* z = (const float*)d_in[0];
  float* out = (float*)d_out;
  const size_t NN = (size_t)NV * NV;

  char* w = (char*)d_ws;
  float*    distS1 = (float*)w;      w += NN * sizeof(float);
  _Float16* maskh  = (_Float16*)w;   w += NN * sizeof(_Float16);
  _Float16* Rh     = (_Float16*)w;   w += NN * sizeof(_Float16);
  _Float16* zh     = (_Float16*)w;   w += (size_t)NV * DV * sizeof(_Float16);
  _Float16* zl     = (_Float16*)w;   w += (size_t)NV * DV * sizeof(_Float16);
  float*    sq     = (float*)w;      w += NV * sizeof(float);
  float*    kth    = (float*)w;      w += NV * sizeof(float);
  float*    msum   = (float*)w;      w += NV * sizeof(float);
  float*    rsum   = (float*)w;      w += NV * sizeof(float);

  prep_kernel<<<NV / 8, 256, 0, stream>>>(z, zh, zl, sq);
  dist_kernel<<<dim3(NV / 64, NV / 128), 256, 0, stream>>>(zh, zl, sq, distS1);
  kth_kernel<<<NV, 256, 0, stream>>>(distS1, kth);
  mask_kernel<<<NV, 256, 0, stream>>>(distS1, kth, maskh, msum);
  r_kernel<<<dim3(NV / 64, NV / 64), 256, 0, stream>>>(maskh, Rh);
  rsum_kernel<<<NV, 256, 0, stream>>>(Rh, rsum);
  s1_kernel<<<dim3(NV / 64, NV / 128), 256, 0, stream>>>(maskh, msum, distS1); // S1 reuses dist
  s2_kernel<<<dim3(NV / 64, NV / 128), 256, 0, stream>>>(distS1, Rh, rsum, out);
  sym_kernel<<<dim3(NV / 64, NV / 64), 256, 0, stream>>>(out);
}
